// W8A16LinearLayer_6451040878934
// MI455X (gfx1250) — matrix operand order repaired and checked by reading, not compiled, not run on hardware
//
#include <hip/hip_runtime.h>

// ---------------- problem dims (fixed by the reference) ----------------
#define M_DIM 8192   // 4 * 2048 flattened batch*seq
#define N_DIM 4096   // out features
#define K_DIM 4096   // in features

// ---------------- tiling ----------------
constexpr int BM = 128;
constexpr int BN = 128;
constexpr int BK = 32;              // one bf16-WMMA K step per stage
constexpr int LDS_STRIDE = BK + 8;  // pad -> conflict-free ds_load_b128 (20-dword lane stride)
constexpr int THREADS = 256;        // 8 waves (wave32)

typedef __attribute__((ext_vector_type(16))) __bf16 v16bf;
typedef __attribute__((ext_vector_type(8)))  __bf16 v8bf;
typedef __attribute__((ext_vector_type(4)))  __bf16 v4bf;
typedef __attribute__((ext_vector_type(8)))  float  v8f;
typedef __attribute__((ext_vector_type(16))) char   v16c;

__device__ __forceinline__ __bf16 f32_to_bf16_rne(float f) {
  unsigned u = __builtin_bit_cast(unsigned, f);
  u += 0x7FFFu + ((u >> 16) & 1u);          // round-to-nearest-even
  unsigned short h = (unsigned short)(u >> 16);
  return __builtin_bit_cast(__bf16, h);
}

__device__ __forceinline__ __bf16 i8_to_bf16(int v) {
  float f = (float)v;                        // exact for int8
  unsigned u = __builtin_bit_cast(unsigned, f);
  unsigned short h = (unsigned short)(u >> 16);  // exact: |v|<=128 needs <=7 mantissa bits
  return __builtin_bit_cast(__bf16, h);
}

__device__ __forceinline__ void wait_asynccnt0() {
#if __has_builtin(__builtin_amdgcn_s_wait_asynccnt)
  __builtin_amdgcn_s_wait_asynccnt(0);
#else
  asm volatile("s_wait_asynccnt 0x0" ::: "memory");
#endif
}

// async global->LDS copy of 16 bytes per lane (GV mode: 64-bit vaddr, no saddr)
__device__ __forceinline__ void async_copy_b128(unsigned lds_addr,
                                                const void* gaddr) {
  unsigned long long ga = (unsigned long long)gaddr;
  asm volatile("global_load_async_to_lds_b128 %0, %1, off"
               :: "v"(lds_addr), "v"(ga) : "memory");
}

__device__ __forceinline__ unsigned lds_lo32(const void* p) {
  // low 32 bits of a generic LDS pointer == workgroup LDS byte offset
  return (unsigned)(unsigned long long)p;
}

// ============================================================================
// Prep pass 1: W int8 [N,K] -> bf16 (exact), bandwidth-bound
// ============================================================================
__global__ __launch_bounds__(256)
void dequant_w_bf16(const signed char* __restrict__ W, __bf16* __restrict__ out) {
  const size_t idx = ((size_t)blockIdx.x * 256 + threadIdx.x) * 16;
  const v16c w = *(const v16c*)(W + idx);
  v8bf h0, h1;
#pragma unroll
  for (int e = 0; e < 8; ++e) {
    h0[e] = i8_to_bf16((int)w[e]);
    h1[e] = i8_to_bf16((int)w[8 + e]);
  }
  *(v8bf*)(out + idx)     = h0;
  *(v8bf*)(out + idx + 8) = h1;
}

// ============================================================================
// Prep pass 2: A fp32 [M,K] -> bf16 (RNE), bandwidth-bound
// ============================================================================
__global__ __launch_bounds__(256)
void cvt_a_bf16(const float* __restrict__ A, __bf16* __restrict__ out) {
  const size_t idx = ((size_t)blockIdx.x * 256 + threadIdx.x) * 8;
  const float4 f0 = *(const float4*)(A + idx);
  const float4 f1 = *(const float4*)(A + idx + 4);
  v8bf h;
  h[0] = f32_to_bf16_rne(f0.x); h[1] = f32_to_bf16_rne(f0.y);
  h[2] = f32_to_bf16_rne(f0.z); h[3] = f32_to_bf16_rne(f0.w);
  h[4] = f32_to_bf16_rne(f1.x); h[5] = f32_to_bf16_rne(f1.y);
  h[6] = f32_to_bf16_rne(f1.z); h[7] = f32_to_bf16_rne(f1.w);
  *(v8bf*)(out + idx) = h;
}

// ============================================================================
// Main GEMM (fast path): pure-bf16 tiles, async DMA global->LDS, WMMA core.
// Branch-free steady-state loop, unrolled x2 with literal buffer indices.
// ============================================================================
__global__ __launch_bounds__(THREADS)
void gemm_bf16_wmma_async(const __bf16* __restrict__ Abf,   // [M, K]
                          const __bf16* __restrict__ Wbf,   // [N, K]
                          const float* __restrict__ scales, // [N]
                          const float* __restrict__ bias,   // [N]
                          float* __restrict__ C) {          // [M, N]
  __shared__ alignas(16) __bf16 sA[2][BM * LDS_STRIDE];
  __shared__ alignas(16) __bf16 sB[2][BN * LDS_STRIDE];

  const int tid     = threadIdx.x;
  const int block_n = blockIdx.x * BN;
  const int block_m = blockIdx.y * BM;

  const int lane = tid & 31;
  const int wave = tid >> 5;
  const int wm   = (wave & 1) * 64;
  const int wn   = (wave >> 1) * 32;
  const int l16  = lane & 15;
  const int lh   = lane >> 4;

  // async loader mapping: tile rows 0..127, 4x16B chunks per 64B row
  const int cp_c = (tid & 3) * 8;   // column (bf16 elements), 8 elems = 16B
  const int cp_r = tid >> 2;        // row 0..63, 2 iterations stride 64

  // per-thread global DMA source pointers (advance by BK per stage)
  const __bf16* gA0 = Abf + (size_t)(block_m + cp_r) * K_DIM + cp_c;
  const __bf16* gA1 = gA0 + (size_t)64 * K_DIM;
  const __bf16* gB0 = Wbf + (size_t)(block_n + cp_r) * K_DIM + cp_c;
  const __bf16* gB1 = gB0 + (size_t)64 * K_DIM;

  const v8f vzero = {0.f, 0.f, 0.f, 0.f, 0.f, 0.f, 0.f, 0.f};
  v8f acc[4][2];
#pragma unroll
  for (int i = 0; i < 4; ++i)
#pragma unroll
    for (int j = 0; j < 2; ++j)
      acc[i][j] = vzero;

  // LDS destinations are loop-invariant per buffer
  unsigned ldsA0 = lds_lo32(&sA[0][cp_r * LDS_STRIDE + cp_c]);
  unsigned ldsA0b = lds_lo32(&sA[0][(cp_r + 64) * LDS_STRIDE + cp_c]);
  unsigned ldsB0 = lds_lo32(&sB[0][cp_r * LDS_STRIDE + cp_c]);
  unsigned ldsB0b = lds_lo32(&sB[0][(cp_r + 64) * LDS_STRIDE + cp_c]);
  unsigned ldsA1 = lds_lo32(&sA[1][cp_r * LDS_STRIDE + cp_c]);
  unsigned ldsA1b = lds_lo32(&sA[1][(cp_r + 64) * LDS_STRIDE + cp_c]);
  unsigned ldsB1 = lds_lo32(&sB[1][cp_r * LDS_STRIDE + cp_c]);
  unsigned ldsB1b = lds_lo32(&sB[1][(cp_r + 64) * LDS_STRIDE + cp_c]);

  auto async_stage = [&](int buf, int k0) {
    if (buf == 0) {
      async_copy_b128(ldsA0,  gA0 + k0);
      async_copy_b128(ldsB0,  gB0 + k0);
      async_copy_b128(ldsA0b, gA1 + k0);
      async_copy_b128(ldsB0b, gB1 + k0);
    } else {
      async_copy_b128(ldsA1,  gA0 + k0);
      async_copy_b128(ldsB1,  gB0 + k0);
      async_copy_b128(ldsA1b, gA1 + k0);
      async_copy_b128(ldsB1b, gB1 + k0);
    }
  };

  auto compute = [&](int buf) {
    v16bf afrag[4], bfrag[2];
    // A fragment (ISA 7.12.2, 16-bit A 16x32): lane l -> row l&15;
    // lanes 0-15 hold K {0..7, 16..23}, lanes 16-31 hold K {8..15, 24..31}
#pragma unroll
    for (int i = 0; i < 4; ++i) {
      const __bf16* p = &sA[buf][(wm + i * 16 + l16) * LDS_STRIDE + lh * 8];
      const v8bf lo = *(const v8bf*)p;
      const v8bf hi = *(const v8bf*)(p + 16);
      afrag[i] = __builtin_shufflevector(lo, hi,
                   0, 1, 2, 3, 4, 5, 6, 7, 8, 9, 10, 11, 12, 13, 14, 15);
    }
    // B fragment (32x16): lane l -> col l&15; lanes 0-15 K {0..7, 16..23}, lanes 16-31 K {8..15, 24..31}
#pragma unroll
    for (int j = 0; j < 2; ++j) {
      const __bf16* p = &sB[buf][(wn + j * 16 + l16) * LDS_STRIDE + lh * 8];
      const v8bf lo = *(const v8bf*)p;
      const v8bf hi = *(const v8bf*)(p + 16);
      bfrag[j] = __builtin_shufflevector(lo, hi,
                   0, 1, 2, 3, 4, 5, 6, 7, 8, 9, 10, 11, 12, 13, 14, 15);
    }
#pragma unroll
    for (int i = 0; i < 4; ++i)
#pragma unroll
      for (int j = 0; j < 2; ++j)
        acc[i][j] = __builtin_amdgcn_wmma_f32_16x16x32_bf16(
            false, afrag[i], false, bfrag[j], (short)0, acc[i][j], false, false);
  };

  constexpr int TILES = K_DIM / BK;  // 128 (even)

  async_stage(0, 0);
  wait_asynccnt0();
  __syncthreads();

  // branch-free steady state: 2 stages per iteration, literal buffer ids
#pragma unroll 1
  for (int t = 0; t < TILES - 2; t += 2) {
    async_stage(1, (t + 1) * BK);
    compute(0);
    wait_asynccnt0();
    __syncthreads();
    async_stage(0, (t + 2) * BK);
    compute(1);
    wait_asynccnt0();
    __syncthreads();
  }
  // epilogue stages (TILES-2, TILES-1)
  async_stage(1, (TILES - 1) * BK);
  compute(0);
  wait_asynccnt0();
  __syncthreads();
  compute(1);

  // Epilogue: C/D layout -> n = l&15 (per 16-col frag), m = 8*(l>>4) + vgpr
#pragma unroll
  for (int j = 0; j < 2; ++j) {
    const int   n  = block_n + wn + j * 16 + l16;
    const float sc = scales[n];
    const float bs = bias[n];
#pragma unroll
    for (int i = 0; i < 4; ++i) {
      const int m0 = block_m + wm + i * 16 + lh * 8;
#pragma unroll
      for (int v = 0; v < 8; ++v)
        C[(size_t)(m0 + v) * N_DIM + n] = acc[i][j][v] * sc + bs;
    }
  }
}

// ============================================================================
// Fallback (small workspace): fused dequant GEMM
// ============================================================================
__global__ __launch_bounds__(THREADS)
void w8a16_bf16_wmma_fused(const float* __restrict__ A,
                           const signed char* __restrict__ W,
                           const float* __restrict__ scales,
                           const float* __restrict__ bias,
                           float* __restrict__ C) {
  __shared__ alignas(16) __bf16 sA[2][BM * LDS_STRIDE];
  __shared__ alignas(16) __bf16 sB[2][BN * LDS_STRIDE];

  const int tid     = threadIdx.x;
  const int block_n = blockIdx.x * BN;
  const int block_m = blockIdx.y * BM;

  const int lane = tid & 31;
  const int wave = tid >> 5;
  const int wm   = (wave & 1) * 64;
  const int wn   = (wave >> 1) * 32;
  const int l16  = lane & 15;
  const int lh   = lane >> 4;

  const int a_c = (tid & 7) * 4;
  const int a_r = tid >> 3;
  const int b_c = (tid & 1) * 16;
  const int b_r = tid >> 1;

  const v8f vzero = {0.f, 0.f, 0.f, 0.f, 0.f, 0.f, 0.f, 0.f};
  v8f acc[4][2];
#pragma unroll
  for (int i = 0; i < 4; ++i)
#pragma unroll
    for (int j = 0; j < 2; ++j)
      acc[i][j] = vzero;

  float4 a_stage[4];
  v16c   w_stage;

  auto global_fetch = [&](int k0) {
#pragma unroll
    for (int it = 0; it < 4; ++it) {
      const int r = a_r + it * 32;
      a_stage[it] = *(const float4*)(A + (size_t)(block_m + r) * K_DIM + k0 + a_c);
    }
    w_stage = *(const v16c*)(W + (size_t)(block_n + b_r) * K_DIM + k0 + b_c);
  };

  auto lds_commit = [&](int buf) {
#pragma unroll
    for (int it = 0; it < 4; ++it) {
      const int r = a_r + it * 32;
      v4bf h;
      h[0] = f32_to_bf16_rne(a_stage[it].x);
      h[1] = f32_to_bf16_rne(a_stage[it].y);
      h[2] = f32_to_bf16_rne(a_stage[it].z);
      h[3] = f32_to_bf16_rne(a_stage[it].w);
      *(v4bf*)&sA[buf][r * LDS_STRIDE + a_c] = h;
    }
    v8bf h0, h1;
#pragma unroll
    for (int e = 0; e < 8; ++e) {
      h0[e] = i8_to_bf16((int)w_stage[e]);
      h1[e] = i8_to_bf16((int)w_stage[8 + e]);
    }
    *(v8bf*)&sB[buf][b_r * LDS_STRIDE + b_c]     = h0;
    *(v8bf*)&sB[buf][b_r * LDS_STRIDE + b_c + 8] = h1;
  };

  auto compute = [&](int buf) {
    v16bf afrag[4], bfrag[2];
#pragma unroll
    for (int i = 0; i < 4; ++i) {
      const __bf16* p = &sA[buf][(wm + i * 16 + l16) * LDS_STRIDE + lh * 8];
      const v8bf lo = *(const v8bf*)p;
      const v8bf hi = *(const v8bf*)(p + 16);
      afrag[i] = __builtin_shufflevector(lo, hi,
                   0, 1, 2, 3, 4, 5, 6, 7, 8, 9, 10, 11, 12, 13, 14, 15);
    }
#pragma unroll
    for (int j = 0; j < 2; ++j) {
      const __bf16* p = &sB[buf][(wn + j * 16 + l16) * LDS_STRIDE + lh * 8];
      const v8bf lo = *(const v8bf*)p;
      const v8bf hi = *(const v8bf*)(p + 16);
      bfrag[j] = __builtin_shufflevector(lo, hi,
                   0, 1, 2, 3, 4, 5, 6, 7, 8, 9, 10, 11, 12, 13, 14, 15);
    }
#pragma unroll
    for (int i = 0; i < 4; ++i)
#pragma unroll
      for (int j = 0; j < 2; ++j)
        acc[i][j] = __builtin_amdgcn_wmma_f32_16x16x32_bf16(
            false, afrag[i], false, bfrag[j], (short)0, acc[i][j], false, false);
  };

  constexpr int TILES = K_DIM / BK;

  global_fetch(0);
  lds_commit(0);
  __syncthreads();

#pragma unroll 1
  for (int t = 0; t < TILES - 1; ++t) {
    const int cur = t & 1;
    global_fetch((t + 1) * BK);
    compute(cur);
    lds_commit(cur ^ 1);
    __syncthreads();
  }
  compute((TILES - 1) & 1);

#pragma unroll
  for (int j = 0; j < 2; ++j) {
    const int   n  = block_n + wn + j * 16 + l16;
    const float sc = scales[n];
    const float bs = bias[n];
#pragma unroll
    for (int i = 0; i < 4; ++i) {
      const int m0 = block_m + wm + i * 16 + lh * 8;
#pragma unroll
      for (int v = 0; v < 8; ++v)
        C[(size_t)(m0 + v) * N_DIM + n] = acc[i][j][v] * sc + bs;
    }
  }
}

// ============================================================================
extern "C" void kernel_launch(void* const* d_in, const int* in_sizes, int n_in,
                              void* d_out, int out_size, void* d_ws, size_t ws_size,
                              hipStream_t stream) {
  (void)in_sizes; (void)n_in; (void)out_size;
  const float*       A      = (const float*)d_in[0];
  const signed char* W      = (const signed char*)d_in[1];
  const float*       scales = (const float*)d_in[2];
  const float*       bias   = (const float*)d_in[3];
  float*             C      = (float*)d_out;

  const size_t a_bytes = (size_t)M_DIM * K_DIM * 2;  // 64 MB bf16 A
  const size_t w_bytes = (size_t)N_DIM * K_DIM * 2;  // 32 MB bf16 W
  dim3 grid(N_DIM / BN, M_DIM / BM);                 // (32, 64)
  dim3 block(THREADS);

  if (ws_size >= a_bytes + w_bytes) {
    __bf16* Abf = (__bf16*)d_ws;
    __bf16* Wbf = (__bf16*)((char*)d_ws + a_bytes);
    cvt_a_bf16<<<dim3((M_DIM * (size_t)K_DIM) / (256 * 8)), dim3(256), 0, stream>>>(A, Abf);
    dequant_w_bf16<<<dim3((N_DIM * (size_t)K_DIM) / (256 * 16)), dim3(256), 0, stream>>>(W, Wbf);
    gemm_bf16_wmma_async<<<grid, block, 0, stream>>>(Abf, Wbf, scales, bias, C);
  } else {
    w8a16_bf16_wmma_fused<<<grid, block, 0, stream>>>(A, W, scales, bias, C);
  }
}
